// SpAdjDropEdge2_12575664242826
// MI455X (gfx1250) — compile-verified, matrix operand order checked by reading, not run on hardware
//
#include <hip/hip_runtime.h>

#define LATDIM   64
#define N_EDGESC 1000000
#define N_HYPERC 128
#define KEEPK    500000
#define EPSV     1e-5f
#define NBINS    16384

typedef __attribute__((ext_vector_type(16))) _Float16 v16h;
typedef __attribute__((ext_vector_type(8)))  float    v8f;

// ---- helpers -------------------------------------------------------------

// K-index of half-pair p for a lane's K-half group (ISA 05_wmma 16-bit A layout:
// lanes 0-15: K = 0..7,16..23 ; lanes 16-31: K = 8..15,24..31)
__device__ __forceinline__ int pk_k(int p, int kh) {
  return (p < 4 ? 2 * p : 16 + 2 * (p - 4)) + kh * 8;
}

__device__ __forceinline__ float leaky(float x) { return x >= 0.f ? x : 0.5f * x; }

__device__ __forceinline__ v8f vzero8() {
  v8f z = {0.f, 0.f, 0.f, 0.f, 0.f, 0.f, 0.f, 0.f};
  return z;
}

__device__ __forceinline__ v8f wmma_f16(v16h a, v16h b, v8f c) {
  // (neg_a, A, neg_b, B, c_mod, C, reuse_a, reuse_b)
  return __builtin_amdgcn_wmma_f32_16x16x32_f16(false, a, false, b, (short)0, c,
                                                false, false);
}

// Build a 16x32 f16 A-fragment from a 16x64 row-major f16 LDS tile.
__device__ __forceinline__ v16h load_afrag(const _Float16* rows, int kbase, int lane) {
  int m = lane & 15, kh = lane >> 4;
  v16h a;
#pragma unroll
  for (int p = 0; p < 8; ++p) {
    int k = kbase + pk_k(p, kh);
    a[2 * p]     = rows[m * 64 + k];
    a[2 * p + 1] = rows[m * 64 + k + 1];
  }
  return a;
}

// Pre-packed B-fragment: lane-major, 16 halves per lane (32B aligned).
__device__ __forceinline__ v16h load_bfrag(const _Float16* base, int fi, int lane) {
  return *(const v16h*)(base + ((fi * 32 + lane) << 4));
}

__device__ __forceinline__ unsigned sort_key_desc(float z) {
  unsigned u = __float_as_uint(z);
  u = (u & 0x80000000u) ? ~u : (u | 0x80000000u);  // ascending-order key
  return ~u;                                        // invert -> descending
}

// ---- kernels -------------------------------------------------------------

__global__ void k_init(unsigned* p, int n) {
  for (int i = blockIdx.x * blockDim.x + threadIdx.x; i < n;
       i += gridDim.x * blockDim.x)
    p[i] = 0u;
}

// Meta-net + fragment packing (one block).
__global__ __launch_bounds__(256) void k_prep(
    const float* uH, const float* iH, const float* mW1, const float* mb1,
    const float* mW2, const float* mb2, const float* l1W, const float* l1b_in,
    const float* l2W, _Float16* fragU, _Float16* fragI, _Float16* fragL1,
    float* ub, float* ibv, float* l1b, float* w2) {
  __shared__ float hmU[64], hmI[64];
  int tid = threadIdx.x;
  if (tid < 64) {
    float su = 0.f, si = 0.f;
    for (int j = 0; j < N_HYPERC; ++j) {
      su += uH[j * 64 + tid];
      si += iH[j * 64 + tid];
    }
    hmU[tid] = su * (1.f / N_HYPERC);
    hmI[tid] = si * (1.f / N_HYPERC);
  }
  __syncthreads();
  // uW / iW  (64x64) -> 8 fragments (2 K-tiles x 4 N-tiles) each
  for (int idx = tid; idx < 8 * 512; idx += 256) {
    int fi = idx >> 9, r = idx & 511;
    int lane = r >> 4, h = r & 15;
    int kt = fi >> 2, nt = fi & 3;
    int kh = lane >> 4, n = nt * 16 + (lane & 15);
    int k = kt * 32 + pk_k(h >> 1, kh) + (h & 1);
    int col = k * 64 + n;
    float wu = mb1[col], wi = mb1[col];
    for (int j = 0; j < 64; ++j) {
      float m1 = mW1[j * 4096 + col];
      wu += hmU[j] * m1;
      wi += hmI[j] * m1;
    }
    fragU[idx] = (_Float16)wu;
    fragI[idx] = (_Float16)wi;
  }
  // lin1_W (128x64) -> 16 fragments (4 K-tiles x 4 N-tiles)
  for (int idx = tid; idx < 16 * 512; idx += 256) {
    int fi = idx >> 9, r = idx & 511;
    int lane = r >> 4, h = r & 15;
    int kt = fi >> 2, nt = fi & 3;
    int kh = lane >> 4, n = nt * 16 + (lane & 15);
    int k = kt * 32 + pk_k(h >> 1, kh) + (h & 1);
    fragL1[idx] = (_Float16)l1W[k * 64 + n];
  }
  if (tid < 64) {
    float bu = mb2[tid], bi = mb2[tid];
    for (int j = 0; j < 64; ++j) {
      bu += hmU[j] * mW2[j * 64 + tid];
      bi += hmI[j] * mW2[j * 64 + tid];
    }
    ub[tid]  = bu;
    ibv[tid] = bi;
    l1b[tid] = l1b_in[tid];
    w2[tid]  = l2W[tid];
  }
}

// Pass A: gathers + GEMM1 + GEMM2 -> per-column BN1 stats.
__global__ __launch_bounds__(128) void k_passA(
    const int* usrs, const int* itms, const float* uKey, const float* iKey,
    const _Float16* fragU, const _Float16* fragI, const _Float16* fragL1,
    const float* ub, const float* ibv, const float* l1b, float* colsum,
    float* colsq) {
  __shared__ _Float16 sU[4][1024];
  __shared__ _Float16 sI[4][1024];
  __shared__ float bcol[64], bsq[64];
  int tid = threadIdx.x;
  int w = tid >> 5, lane = tid & 31;
  int kh = lane >> 4, n0 = lane & 15;
  long eb = ((long)blockIdx.x * 4 + w) * 16;
  if (tid < 64) { bcol[tid] = 0.f; bsq[tid] = 0.f; }
  {  // gather 16 rows x 64 f32, convert to f16 in LDS
    int m = lane >> 1, kb = (lane & 1) * 32;
    long ru = (long)usrs[eb + m] * 64 + kb;
    long ri = (long)itms[eb + m] * 64 + kb;
#pragma unroll
    for (int t = 0; t < 8; ++t) {
      float4 vu = *(const float4*)(uKey + ru + t * 4);
      float4 vi = *(const float4*)(iKey + ri + t * 4);
      _Float16* du = &sU[w][m * 64 + kb + t * 4];
      du[0] = (_Float16)vu.x; du[1] = (_Float16)vu.y;
      du[2] = (_Float16)vu.z; du[3] = (_Float16)vu.w;
      _Float16* di = &sI[w][m * 64 + kb + t * 4];
      di[0] = (_Float16)vi.x; di[1] = (_Float16)vi.y;
      di[2] = (_Float16)vi.z; di[3] = (_Float16)vi.w;
    }
  }
  __syncthreads();
  // GEMM1: ulat/ilat = key @ W   (16x64x64, 8 WMMA each)
  v8f accU[4], accI[4];
#pragma unroll
  for (int nt = 0; nt < 4; ++nt) { accU[nt] = vzero8(); accI[nt] = vzero8(); }
#pragma unroll
  for (int kt = 0; kt < 2; ++kt) {
    v16h aU = load_afrag(sU[w], kt * 32, lane);
    v16h aI = load_afrag(sI[w], kt * 32, lane);
#pragma unroll
    for (int nt = 0; nt < 4; ++nt) {
      accU[nt] = wmma_f16(aU, load_bfrag(fragU, kt * 4 + nt, lane), accU[nt]);
      accI[nt] = wmma_f16(aI, load_bfrag(fragI, kt * 4 + nt, lane), accI[nt]);
    }
  }
  __syncthreads();
  // bias + leaky, restage lat (f16) for GEMM2
#pragma unroll
  for (int nt = 0; nt < 4; ++nt) {
    int n = nt * 16 + n0;
    float bu = ub[n], bi = ibv[n];
#pragma unroll
    for (int v = 0; v < 8; ++v) {
      int m = v + 8 * kh;
      sU[w][m * 64 + n] = (_Float16)leaky(accU[nt][v] + bu);
      sI[w][m * 64 + n] = (_Float16)leaky(accI[nt][v] + bi);
    }
  }
  __syncthreads();
  // GEMM2: y = [ulat|ilat] @ lin1  (16x128x64, 16 WMMA)
  v8f accY[4];
#pragma unroll
  for (int nt = 0; nt < 4; ++nt) accY[nt] = vzero8();
#pragma unroll
  for (int kt = 0; kt < 4; ++kt) {
    const _Float16* src = (kt < 2) ? sU[w] : sI[w];
    v16h a = load_afrag(src, (kt & 1) * 32, lane);
#pragma unroll
    for (int nt = 0; nt < 4; ++nt)
      accY[nt] = wmma_f16(a, load_bfrag(fragL1, kt * 4 + nt, lane), accY[nt]);
  }
  // column sums for BN1
#pragma unroll
  for (int nt = 0; nt < 4; ++nt) {
    int n = nt * 16 + n0;
    float bb = l1b[n];
    float ls = 0.f, lq = 0.f;
#pragma unroll
    for (int v = 0; v < 8; ++v) {
      float yv = accY[nt][v] + bb;
      ls += yv;
      lq += yv * yv;
    }
    atomicAdd(&bcol[n], ls);
    atomicAdd(&bsq[n], lq);
  }
  __syncthreads();
  if (tid < 64) {
    atomicAdd(&colsum[tid], bcol[tid]);
    atomicAdd(&colsq[tid], bsq[tid]);
  }
}

__global__ void k_fin1(const float* colsum, const float* colsq, float* mu1,
                       float* rs1) {
  int n = threadIdx.x;
  if (n < 64) {
    float mu  = colsum[n] * (1.f / N_EDGESC);
    float var = colsq[n] * (1.f / N_EDGESC) - mu * mu;
    mu1[n] = mu;
    rs1[n] = rsqrtf(var + EPSV);
  }
}

// Pass B: recompute, apply BN1, fold +ulat+ilat, dot lin2 -> pre-BN2 scalar.
__global__ __launch_bounds__(128) void k_passB(
    const int* usrs, const int* itms, const float* uKey, const float* iKey,
    const _Float16* fragU, const _Float16* fragI, const _Float16* fragL1,
    const float* ub, const float* ibv, const float* l1b, const float* mu1,
    const float* rs1, const float* g1, const float* be1, const float* w2,
    const float* l2b, float* sPre, float* gss) {
  __shared__ _Float16 sU[4][1024];
  __shared__ _Float16 sI[4][1024];
  __shared__ float swacc[4][16];
  __shared__ float bs[2];
  int tid = threadIdx.x;
  int w = tid >> 5, lane = tid & 31;
  int kh = lane >> 4, n0 = lane & 15;
  long eb = ((long)blockIdx.x * 4 + w) * 16;
  if (lane < 16) swacc[w][lane] = 0.f;
  if (tid < 2) bs[tid] = 0.f;
  {
    int m = lane >> 1, kb = (lane & 1) * 32;
    long ru = (long)usrs[eb + m] * 64 + kb;
    long ri = (long)itms[eb + m] * 64 + kb;
#pragma unroll
    for (int t = 0; t < 8; ++t) {
      float4 vu = *(const float4*)(uKey + ru + t * 4);
      float4 vi = *(const float4*)(iKey + ri + t * 4);
      _Float16* du = &sU[w][m * 64 + kb + t * 4];
      du[0] = (_Float16)vu.x; du[1] = (_Float16)vu.y;
      du[2] = (_Float16)vu.z; du[3] = (_Float16)vu.w;
      _Float16* di = &sI[w][m * 64 + kb + t * 4];
      di[0] = (_Float16)vi.x; di[1] = (_Float16)vi.y;
      di[2] = (_Float16)vi.z; di[3] = (_Float16)vi.w;
    }
  }
  __syncthreads();
  v8f accU[4], accI[4];
#pragma unroll
  for (int nt = 0; nt < 4; ++nt) { accU[nt] = vzero8(); accI[nt] = vzero8(); }
#pragma unroll
  for (int kt = 0; kt < 2; ++kt) {
    v16h aU = load_afrag(sU[w], kt * 32, lane);
    v16h aI = load_afrag(sI[w], kt * 32, lane);
#pragma unroll
    for (int nt = 0; nt < 4; ++nt) {
      accU[nt] = wmma_f16(aU, load_bfrag(fragU, kt * 4 + nt, lane), accU[nt]);
      accI[nt] = wmma_f16(aI, load_bfrag(fragI, kt * 4 + nt, lane), accI[nt]);
    }
  }
  __syncthreads();
#pragma unroll
  for (int nt = 0; nt < 4; ++nt) {
    int n = nt * 16 + n0;
    float bu = ub[n], bi = ibv[n];
#pragma unroll
    for (int v = 0; v < 8; ++v) {
      int m = v + 8 * kh;
      sU[w][m * 64 + n] = (_Float16)leaky(accU[nt][v] + bu);
      sI[w][m * 64 + n] = (_Float16)leaky(accI[nt][v] + bi);
    }
  }
  __syncthreads();
  v8f accY[4];
#pragma unroll
  for (int nt = 0; nt < 4; ++nt) accY[nt] = vzero8();
#pragma unroll
  for (int kt = 0; kt < 4; ++kt) {
    const _Float16* src = (kt < 2) ? sU[w] : sI[w];
    v16h a = load_afrag(src, (kt & 1) * 32, lane);
#pragma unroll
    for (int nt = 0; nt < 4; ++nt)
      accY[nt] = wmma_f16(a, load_bfrag(fragL1, kt * 4 + nt, lane), accY[nt]);
  }
  // h = leaky(bn1(y)); per-edge dot with w2 via LDS float atomics
#pragma unroll
  for (int nt = 0; nt < 4; ++nt) {
    int n = nt * 16 + n0;
    float bb = l1b[n], mu = mu1[n], rs = rs1[n], g = g1[n], be = be1[n],
          wn = w2[n];
#pragma unroll
    for (int v = 0; v < 8; ++v) {
      int m = v + 8 * kh;
      float yv = accY[nt][v] + bb;
      float h = leaky((yv - mu) * rs * g + be);
      atomicAdd(&swacc[w][m], h * wn);
    }
  }
  // q = (ulat+ilat) @ w2 from the staged lat
  float q = 0.f;
  if (lane < 16) {
    int m = lane;
    for (int n = 0; n < 64; ++n)
      q += ((float)sU[w][m * 64 + n] + (float)sI[w][m * 64 + n]) * w2[n];
  }
  __syncthreads();
  if (lane < 16) {
    float s = swacc[w][lane] + q + l2b[0];
    sPre[eb + lane] = s;
    atomicAdd(&bs[0], s);
    atomicAdd(&bs[1], s * s);
  }
  __syncthreads();
  if (tid < 2) atomicAdd(&gss[tid], bs[tid]);
}

__global__ void k_fin2(float* ss) {
  if (threadIdx.x == 0) {
    float mu  = ss[0] * (1.f / N_EDGESC);
    float var = ss[1] * (1.f / N_EDGESC) - mu * mu;
    ss[2] = mu;
    ss[3] = rsqrtf(var + EPSV);
  }
}

__global__ void k_hist(const float* sPre, const float* ss, const float* g2,
                       const float* b2, unsigned* hist) {
  float mu = ss[2], rs = ss[3], g = g2[0], b = b2[0];
  for (int e = blockIdx.x * blockDim.x + threadIdx.x; e < N_EDGESC;
       e += gridDim.x * blockDim.x) {
    float z = (sPre[e] - mu) * rs * g + b;  // sigmoid is monotonic: rank on z
    atomicAdd(&hist[sort_key_desc(z) >> 18], 1u);
  }
}

__global__ __launch_bounds__(256) void k_scan(const unsigned* hist,
                                              unsigned* base) {
  __shared__ unsigned part[256];
  int tid = threadIdx.x;
  unsigned s = 0;
  for (int j = 0; j < NBINS / 256; ++j) s += hist[tid * (NBINS / 256) + j];
  part[tid] = s;
  __syncthreads();
  for (int off = 1; off < 256; off <<= 1) {
    unsigned v = (tid >= off) ? part[tid - off] : 0u;
    __syncthreads();
    part[tid] += v;
    __syncthreads();
  }
  unsigned run = (tid == 0) ? 0u : part[tid - 1];
  for (int j = 0; j < NBINS / 256; ++j) {
    unsigned h = hist[tid * (NBINS / 256) + j];
    base[tid * (NBINS / 256) + j] = run;
    run += h;
  }
}

__global__ void k_scatter(const float* sPre, const float* ss, const float* g2,
                          const float* b2, const unsigned* base, unsigned* cnt,
                          unsigned* order) {
  float mu = ss[2], rs = ss[3], g = g2[0], b = b2[0];
  for (int e = blockIdx.x * blockDim.x + threadIdx.x; e < N_EDGESC;
       e += gridDim.x * blockDim.x) {
    float z = (sPre[e] - mu) * rs * g + b;
    unsigned bin = sort_key_desc(z) >> 18;
    unsigned r = base[bin] + atomicAdd(&cnt[bin], 1u);
    if (r < KEEPK) order[r] = (unsigned)e;
  }
}

__global__ void k_emit(const unsigned* order, const int* adj_idx,
                       const float* adj_vals, int* iout, float* fout) {
  for (int r = blockIdx.x * blockDim.x + threadIdx.x; r < KEEPK;
       r += gridDim.x * blockDim.x) {
    int e = (int)order[r];
    iout[r]             = adj_idx[e];
    iout[KEEPK + r]     = adj_idx[N_EDGESC + e];
    fout[2 * KEEPK + r] = adj_vals[e];
  }
}

// ---- launch --------------------------------------------------------------

extern "C" void kernel_launch(void* const* d_in, const int* in_sizes, int n_in,
                              void* d_out, int out_size, void* d_ws,
                              size_t ws_size, hipStream_t stream) {
  (void)in_sizes; (void)n_in; (void)out_size; (void)ws_size;
  const int*   usrs = (const int*)d_in[0];
  const int*   itms = (const int*)d_in[1];
  const float* uKey = (const float*)d_in[2];
  const float* iKey = (const float*)d_in[3];
  const float* uH   = (const float*)d_in[4];
  const float* iH   = (const float*)d_in[5];
  const float* adjV = (const float*)d_in[6];
  const int*   adjI = (const int*)d_in[7];
  const float* mW1  = (const float*)d_in[8];
  const float* mb1  = (const float*)d_in[9];
  const float* mW2  = (const float*)d_in[10];
  const float* mb2  = (const float*)d_in[11];
  const float* l1W  = (const float*)d_in[12];
  const float* l1bi = (const float*)d_in[13];
  const float* l2W  = (const float*)d_in[14];
  const float* l2b  = (const float*)d_in[15];
  const float* g1   = (const float*)d_in[16];
  const float* be1  = (const float*)d_in[17];
  const float* g2   = (const float*)d_in[18];
  const float* b2   = (const float*)d_in[19];

  char* ws = (char*)d_ws;
  _Float16* fragU  = (_Float16*)(ws + 0);        // 8 frags * 1KB
  _Float16* fragI  = (_Float16*)(ws + 8192);     // 8 frags
  _Float16* fragL1 = (_Float16*)(ws + 16384);    // 16 frags
  float* ub     = (float*)(ws + 32768);
  float* ibv    = (float*)(ws + 33024);
  float* l1b    = (float*)(ws + 33280);
  float* w2     = (float*)(ws + 33536);
  float* colsum = (float*)(ws + 33792);
  float* colsq  = (float*)(ws + 34048);
  float* mu1    = (float*)(ws + 34304);
  float* rs1    = (float*)(ws + 34560);
  float* ss     = (float*)(ws + 34816);          // ssum, ssq, mu2, rs2
  unsigned* hist  = (unsigned*)(ws + 35072);     // 16384 u32
  unsigned* base  = (unsigned*)(ws + 100608);    // 16384 u32
  unsigned* cnt   = (unsigned*)(ws + 166144);    // 16384 u32
  float* sPre     = (float*)(ws + 231680);       // 1e6 f32
  unsigned* order = (unsigned*)(ws + 4231680);   // 500k u32

  // zero all accumulators/histograms: bytes [33792, 231680) = 49472 dwords
  k_init<<<194, 256, 0, stream>>>((unsigned*)(ws + 33792), 49472);
  k_prep<<<1, 256, 0, stream>>>(uH, iH, mW1, mb1, mW2, mb2, l1W, l1bi, l2W,
                                fragU, fragI, fragL1, ub, ibv, l1b, w2);
  k_passA<<<15625, 128, 0, stream>>>(usrs, itms, uKey, iKey, fragU, fragI,
                                     fragL1, ub, ibv, l1b, colsum, colsq);
  k_fin1<<<1, 64, 0, stream>>>(colsum, colsq, mu1, rs1);
  k_passB<<<15625, 128, 0, stream>>>(usrs, itms, uKey, iKey, fragU, fragI,
                                     fragL1, ub, ibv, l1b, mu1, rs1, g1, be1,
                                     w2, l2b, sPre, ss);
  k_fin2<<<1, 1, 0, stream>>>(ss);
  k_hist<<<2048, 256, 0, stream>>>(sPre, ss, g2, b2, hist);
  k_scan<<<1, 256, 0, stream>>>(hist, base);
  k_scatter<<<2048, 256, 0, stream>>>(sPre, ss, g2, b2, base, cnt, order);
  k_emit<<<1024, 256, 0, stream>>>(order, adjI, adjV, (int*)d_out,
                                   (float*)d_out);
}
